// LearnableMultiheadSelfAttention_42210938585581
// MI455X (gfx1250) — matrix operand order repaired and checked by reading, not compiled, not run on hardware
//
#include <hip/hip_runtime.h>

// ---------------------------------------------------------------------------
// Problem constants (match reference)
// ---------------------------------------------------------------------------
constexpr int kT  = 512;
constexpr int kB  = 4;
constexpr int kD  = 512;
constexpr int kH  = 8;
constexpr int kE  = 64;
constexpr int kM  = 4;
constexpr int kNL = 512;
constexpr int kL  = kM * kT + kNL + kT;   // 3072
constexpr float kScale = 0.125f;          // 1/sqrt(64)

// ---------------------------------------------------------------------------
// WMMA types & helpers (CDNA5 / gfx1250, wave32)
// ---------------------------------------------------------------------------
typedef __attribute__((ext_vector_type(16))) __bf16 bf16x16;
typedef __attribute__((ext_vector_type(8)))  float  f32x8;

union FragAB { uint4 q[2]; bf16x16 v; };
union FragC  { f32x8 v; float f[8]; };

__device__ __forceinline__ f32x8 wmma_bf16(bf16x16 a, bf16x16 b, f32x8 c) {
  // v_wmma_f32_16x16x32_bf16  D = A(16x32) * B(32x16) + C
  return __builtin_amdgcn_wmma_f32_16x16x32_bf16(false, a, false, b,
                                                 (short)0, c, false, false);
}

// Async global -> LDS copy, 16 bytes per lane (CDNA5, tracked by ASYNCcnt).
// lds_byte_off is the LDS byte offset (dynamic-LDS base == 0 when there is
// no static LDS in the kernel).
__device__ __forceinline__ void async_b128(unsigned lds_byte_off,
                                           const void* gptr) {
  asm volatile("global_load_async_to_lds_b128 %0, %1, off"
               :: "v"(lds_byte_off), "v"(gptr)
               : "memory");
}
__device__ __forceinline__ void async_wait0() {
  asm volatile("s_wait_asynccnt 0x0" ::: "memory");
}

// A fragment: 16x32 bf16 tile from LDS, rows contiguous, strideQ = row stride
// in uint4 units, kQ = k-chunk offset in uint4 units.
// lanes 0-15: row M=lane, K0-7 (q0) + K16-23 (q1); lanes 16-31: K8-15 + K24-31
__device__ __forceinline__ bf16x16 load_fragA(const uint4* lds, int rbase,
                                              int strideQ, int kQ, int lane) {
  int half = (lane >> 4) & 1;
  int r = rbase + (lane & 15);
  FragAB u;
  u.q[0] = lds[r * strideQ + kQ + half];
  u.q[1] = lds[r * strideQ + kQ + 2 + half];
  return u.v;
}

// B fragment: 32x16 bf16, stored in LDS as one row per B-column (B^T rows).
// lanes 0-15: col N=lane, K0-7 + K16-23; lanes 16-31: col N=lane-16, K8-15 + K24-31
__device__ __forceinline__ bf16x16 load_fragB(const uint4* lds, int nbase,
                                              int strideQ, int kQ, int lane) {
  int half = (lane >> 4) & 1;
  int n = nbase + (lane & 15);
  FragAB u;
  u.q[0] = lds[n * strideQ + kQ + half];
  u.q[1] = lds[n * strideQ + kQ + 2 + half];
  return u.v;
}

// ---------------------------------------------------------------------------
// Elementwise prep kernels
// ---------------------------------------------------------------------------
__global__ void build_cbf_kernel(const float* __restrict__ mem,
                                 const float* __restrict__ x,
                                 __bf16* __restrict__ cbf) {
  size_t idx = (size_t)blockIdx.x * 256 + threadIdx.x;  // < 10240*512
  const size_t memN = (size_t)kM * kT * kB * kD;        // 4,194,304
  float v = (idx < memN) ? mem[idx] : x[idx - memN];
  cbf[idx] = (__bf16)v;
}

__global__ void f2bf_kernel(const float* __restrict__ src,
                            __bf16* __restrict__ dst, size_t n) {
  size_t idx = (size_t)blockIdx.x * 256 + threadIdx.x;
  if (idx < n) dst[idx] = (__bf16)src[idx];
}

// WT[n*rows + d] = W[d*cols + n]   (W is rows x cols, row-major)
__global__ void transpose_bf_kernel(const float* __restrict__ W,
                                    __bf16* __restrict__ WT,
                                    int rows, int cols) {
  size_t idx = (size_t)blockIdx.x * 256 + threadIdx.x;
  if (idx >= (size_t)rows * cols) return;
  int d = (int)(idx / cols);
  int n = (int)(idx % cols);
  WT[(size_t)n * rows + d] = (__bf16)W[idx];
}

// QU/QV: [b][h][t][e] bf16, q + pos_bias
__global__ void build_qubf_kernel(const float* __restrict__ Qf,
                                  const float* __restrict__ pbu,
                                  const float* __restrict__ pbv,
                                  __bf16* __restrict__ QU,
                                  __bf16* __restrict__ QV) {
  size_t idx = (size_t)blockIdx.x * 256 + threadIdx.x;  // < B*H*T*E
  int e = (int)(idx & 63);
  int t = (int)((idx >> 6) & 511);
  int h = (int)((idx >> 15) & 7);
  int b = (int)(idx >> 18);
  float q = Qf[((size_t)t * kB + b) * kD + h * kE + e];
  QU[idx] = (__bf16)(q + pbu[h * kE + e]);
  QV[idx] = (__bf16)(q + pbv[h * kE + e]);
}

// Kcat/Rcat: [b][h][col<kL][e] bf16 — concatenated key / rel rows
__global__ void build_kcat_kernel(const float* __restrict__ KV,
                                  const float* __restrict__ NKV,
                                  const float* __restrict__ REL,
                                  __bf16* __restrict__ Kcat,
                                  __bf16* __restrict__ Rcat) {
  size_t idx = (size_t)blockIdx.x * 256 + threadIdx.x;  // < B*H*L*E
  int e   = (int)(idx & 63);
  int col = (int)((idx >> 6) % kL);
  int bh  = (int)(idx / ((size_t)kL * kE));
  int b = bh >> 3, h = bh & 7;
  float kval;
  if (col < kM * kT)
    kval = KV[((size_t)col * kB + b) * (2 * kD) + h * kE + e];
  else if (col < kM * kT + kNL)
    kval = NKV[((size_t)(col - kM * kT) * kB + b) * (2 * kD) + h * kE + e];
  else
    kval = KV[((size_t)(col - kNL) * kB + b) * (2 * kD) + h * kE + e];
  Kcat[idx] = (__bf16)kval;
  // rel rows are already in concatenated order: cache | nei | inp
  Rcat[idx] = (__bf16)REL[((size_t)col * kB + b) * kD + h * kE + e];
}

// Vcat: value rows scaled by weights per segment (nei unscaled)
__global__ void build_vcat_kernel(const float* __restrict__ KV,
                                  const float* __restrict__ NKV,
                                  const float* __restrict__ wts,
                                  __bf16* __restrict__ Vcat) {
  size_t idx = (size_t)blockIdx.x * 256 + threadIdx.x;  // < B*H*L*E
  int e   = (int)(idx & 63);
  int col = (int)((idx >> 6) % kL);
  int bh  = (int)(idx / ((size_t)kL * kE));
  int b = bh >> 3, h = bh & 7;
  float v, w = 1.f;
  if (col < kM * kT) {
    int seg = col >> 9, j = col & 511;
    v = KV[((size_t)col * kB + b) * (2 * kD) + kD + h * kE + e];
    w = wts[((size_t)j * kB + b) * (kM + 1) + seg];
  } else if (col < kM * kT + kNL) {
    v = NKV[((size_t)(col - kM * kT) * kB + b) * (2 * kD) + kD + h * kE + e];
  } else {
    int j = col - (kM * kT + kNL);
    v = KV[((size_t)(col - kNL) * kB + b) * (2 * kD) + kD + h * kE + e];
    w = wts[((size_t)j * kB + b) * (kM + 1) + kM];
  }
  Vcat[idx] = (__bf16)(v * w);
}

// ---------------------------------------------------------------------------
// Generic bf16 WMMA GEMM:  C[N x Ncols] = A[N x Kdim] * Bt[Ncols x Kdim]^T
// Block: 256 thr = 8 waves, tile 128x128, K-chunks of 32.
// LDS staged with async global->LDS copies (ASYNCcnt).
// Dynamic LDS: [0,8192) = A tile, [8192,16384) = B tile.
// ---------------------------------------------------------------------------
__global__ __launch_bounds__(256)
void gemm_bf16_kernel(const __bf16* __restrict__ A,
                      const __bf16* __restrict__ Bt,
                      float* __restrict__ C, int Kdim, int Ncols) {
  extern __shared__ char smem[];
  uint4* As4 = (uint4*)smem;            // 128 rows x 32 bf16
  uint4* Bs4 = (uint4*)(smem + 8192);   // 128 cols x 32 bf16
  const int tid = threadIdx.x, lane = tid & 31, wv = tid >> 5;
  const int rowBase = blockIdx.x * 128;
  const int colBase = blockIdx.y * 128;
  const int wr = wv >> 1;  // 0..3 -> 32-row group
  const int wc = wv & 1;   // 0..1 -> 64-col group

  FragC acc[2][4];
  for (int i = 0; i < 2; ++i)
    for (int j = 0; j < 4; ++j)
      for (int g = 0; g < 8; ++g) acc[i][j].f[g] = 0.f;

  const int nK = Kdim >> 5;
  for (int kk = 0; kk < nK; ++kk) {
    for (int it = 0; it < 2; ++it) {
      int idx = tid + it * 256;
      int row = idx >> 2, q = idx & 3;
      async_b128((unsigned)(idx * 16),
                 A + (size_t)(rowBase + row) * Kdim + kk * 32 + q * 8);
      async_b128((unsigned)(8192 + idx * 16),
                 Bt + (size_t)(colBase + row) * Kdim + kk * 32 + q * 8);
    }
    async_wait0();
    __syncthreads();
    bf16x16 a0 = load_fragA(As4, wr * 32 + 0,  4, 0, lane);
    bf16x16 a1 = load_fragA(As4, wr * 32 + 16, 4, 0, lane);
    for (int ct = 0; ct < 4; ++ct) {
      bf16x16 b = load_fragB(Bs4, wc * 64 + ct * 16, 4, 0, lane);
      acc[0][ct].v = wmma_bf16(a0, b, acc[0][ct].v);
      acc[1][ct].v = wmma_bf16(a1, b, acc[1][ct].v);
    }
    __syncthreads();
  }

  const int half = lane >> 4, n = lane & 15;
  for (int rt = 0; rt < 2; ++rt)
    for (int ct = 0; ct < 4; ++ct)
      for (int g = 0; g < 8; ++g) {
        int r = rowBase + wr * 32 + rt * 16 + g + 8 * half;
        int c = colBase + wc * 64 + ct * 16 + n;
        C[(size_t)r * Ncols + c] = acc[rt][ct].f[g];
      }
}

// ---------------------------------------------------------------------------
// Score kernels: per (b,h) plane, S = Q'(T x 64) * Kcat^T(64 x L)
// Block: 256 thr, tile 64 rows x 256 cols, full K=64 staged once (async).
// Dynamic LDS: [0,8192) = Q tile (64x64), [8192,40960) = K tile (256x64).
// ---------------------------------------------------------------------------
__global__ __launch_bounds__(256)
void score_ac_kernel(const __bf16* __restrict__ QU,
                     const __bf16* __restrict__ Kcat,
                     const float* __restrict__ indice,
                     float* __restrict__ score) {
  extern __shared__ char smem[];
  uint4* As4 = (uint4*)smem;            // 64 x 64 bf16
  uint4* Bs4 = (uint4*)(smem + 8192);   // 256 x 64 bf16
  const int tid = threadIdx.x, lane = tid & 31, wv = tid >> 5;
  const int bh = blockIdx.z, b = bh >> 3;
  const int rowBase = blockIdx.y * 64;
  const int colBase = blockIdx.x * 256;

  for (int it = 0; it < 2; ++it) {
    int idx = tid + it * 256, row = idx >> 3, q = idx & 7;
    async_b128((unsigned)(idx * 16),
               QU + ((size_t)bh * kT + rowBase + row) * kE + q * 8);
  }
  for (int it = 0; it < 8; ++it) {
    int idx = tid + it * 256, row = idx >> 3, q = idx & 7;
    async_b128((unsigned)(8192 + idx * 16),
               Kcat + ((size_t)bh * kL + colBase + row) * kE + q * 8);
  }
  async_wait0();
  __syncthreads();

  const int rt = wv & 3;   // row tile (16 rows)
  const int cg = wv >> 2;  // col half (128 cols)
  FragC acc[8];
  for (int ct = 0; ct < 8; ++ct)
    for (int g = 0; g < 8; ++g) acc[ct].f[g] = 0.f;

  for (int ks = 0; ks < 2; ++ks) {
    bf16x16 a = load_fragA(As4, rt * 16, 8, ks * 4, lane);
    for (int ct = 0; ct < 8; ++ct) {
      bf16x16 bfr = load_fragB(Bs4, cg * 128 + ct * 16, 8, ks * 4, lane);
      acc[ct].v = wmma_bf16(a, bfr, acc[ct].v);
    }
  }

  const int half = lane >> 4, n = lane & 15;
  for (int ct = 0; ct < 8; ++ct)
    for (int g = 0; g < 8; ++g) {
      int i   = rowBase + rt * 16 + g + 8 * half;
      int col = colBase + cg * 128 + ct * 16 + n;
      int seg = (col < kM * kT) ? (col >> 9) : ((col < kM * kT + kNL) ? -1 : kM);
      float f = (seg < 0) ? 1.f : indice[((size_t)i * kB + b) * (kM + 1) + seg];
      score[((size_t)bh * kT + i) * kL + col] = acc[ct].f[g] * f;
    }
}

// BD: same GEMM, but fold the rel_shift into a scatter-add epilogue.
// BD[i][c] lands at padded-flat p = i*(L+1)+c+1; out flat t = p - T.
__global__ __launch_bounds__(256)
void score_bd_kernel(const __bf16* __restrict__ QV,
                     const __bf16* __restrict__ Rcat,
                     const float* __restrict__ indice,
                     float* __restrict__ score) {
  extern __shared__ char smem[];
  uint4* As4 = (uint4*)smem;
  uint4* Bs4 = (uint4*)(smem + 8192);
  const int tid = threadIdx.x, lane = tid & 31, wv = tid >> 5;
  const int bh = blockIdx.z, b = bh >> 3;
  const int rowBase = blockIdx.y * 64;
  const int colBase = blockIdx.x * 256;

  for (int it = 0; it < 2; ++it) {
    int idx = tid + it * 256, row = idx >> 3, q = idx & 7;
    async_b128((unsigned)(idx * 16),
               QV + ((size_t)bh * kT + rowBase + row) * kE + q * 8);
  }
  for (int it = 0; it < 8; ++it) {
    int idx = tid + it * 256, row = idx >> 3, q = idx & 7;
    async_b128((unsigned)(8192 + idx * 16),
               Rcat + ((size_t)bh * kL + colBase + row) * kE + q * 8);
  }
  async_wait0();
  __syncthreads();

  const int rt = wv & 3, cg = wv >> 2;
  FragC acc[8];
  for (int ct = 0; ct < 8; ++ct)
    for (int g = 0; g < 8; ++g) acc[ct].f[g] = 0.f;

  for (int ks = 0; ks < 2; ++ks) {
    bf16x16 a = load_fragA(As4, rt * 16, 8, ks * 4, lane);
    for (int ct = 0; ct < 8; ++ct) {
      bf16x16 bfr = load_fragB(Bs4, cg * 128 + ct * 16, 8, ks * 4, lane);
      acc[ct].v = wmma_bf16(a, bfr, acc[ct].v);
    }
  }

  const int half = lane >> 4, n = lane & 15;
  float* plane = score + (size_t)bh * kT * kL;
  for (int ct = 0; ct < 8; ++ct)
    for (int g = 0; g < 8; ++g) {
      int i   = rowBase + rt * 16 + g + 8 * half;
      int col = colBase + cg * 128 + ct * 16 + n;
      int seg = (col < kM * kT) ? (col >> 9) : ((col < kM * kT + kNL) ? -1 : kM);
      float f = (seg < 0) ? 1.f : indice[((size_t)i * kB + b) * (kM + 1) + seg];
      float val = acc[ct].f[g] * f;
      long long t = (long long)i * (kL + 1) + col + 1 - kT;
      if (t >= 0 && t < (long long)kT * kL) {
        size_t o = (size_t)t;            // == i2*kL + j2
        plane[o] += val;                 // unique destination per (i,col)
      }
    }
}

// ---------------------------------------------------------------------------
// Softmax over j (L=3072) per (bh, i) row; scale + mask applied here.
// prob written back in place.
// ---------------------------------------------------------------------------
__global__ __launch_bounds__(256)
void softmax_kernel(float* __restrict__ score,
                    const unsigned char* __restrict__ mask) {
  const int rowId = blockIdx.x;            // < B*H*T
  const int bh = rowId / kT, i = rowId % kT;
  const int b = bh >> 3;
  float* row = score + ((size_t)bh * kT + i) * kL;
  const unsigned char* mrow = mask + ((size_t)i * kL) * kB + b;
  __shared__ float red[256];
  const int tid = threadIdx.x;

  float vals[12];
  float lmax = -__builtin_inff();
  for (int k = 0; k < 12; ++k) {
    int j = tid + k * 256;
    float v = row[j] * kScale;
    if (mrow[(size_t)j * kB]) v = -__builtin_inff();
    vals[k] = v;
    lmax = fmaxf(lmax, v);
  }
  red[tid] = lmax; __syncthreads();
  for (int s = 128; s > 0; s >>= 1) {
    if (tid < s) red[tid] = fmaxf(red[tid], red[tid + s]);
    __syncthreads();
  }
  float m = red[0]; __syncthreads();

  float lsum = 0.f;
  for (int k = 0; k < 12; ++k) {
    float e = __expf(vals[k] - m);
    vals[k] = e; lsum += e;
  }
  red[tid] = lsum; __syncthreads();
  for (int s = 128; s > 0; s >>= 1) {
    if (tid < s) red[tid] += red[tid + s];
    __syncthreads();
  }
  float inv = 1.f / red[0];
  for (int k = 0; k < 12; ++k) row[tid + k * 256] = vals[k] * inv;
}

// attn_matrix = mean over (b,h)
__global__ void attnmat_kernel(const float* __restrict__ score,
                               float* __restrict__ out2) {
  size_t idx = (size_t)blockIdx.x * 256 + threadIdx.x;  // < T*L
  if (idx >= (size_t)kT * kL) return;
  float s = 0.f;
  for (int bh = 0; bh < kB * kH; ++bh)
    s += score[(size_t)bh * kT * kL + idx];
  out2[idx] = s * (1.f / (kB * kH));
}

// ---------------------------------------------------------------------------
// PV: per (b,h) plane, vec(T x 64) = P(T x L) * Vcat(L x 64)
// Block: 256 thr = 8 waves, tile 128 rows x 64 cols, K chunks of 32.
// Prob (f32) converted to bf16 during LDS staging (sync path: conversion
// and transpose can't use the async copy engine).
// ---------------------------------------------------------------------------
__global__ __launch_bounds__(256)
void pv_kernel(const float* __restrict__ P,
               const __bf16* __restrict__ Vcat,
               float* __restrict__ VECf) {
  __shared__ uint4 As4[512];  // 128 x 32 bf16
  __shared__ uint4 Bs4[256];  // 64 (e) x 32 (k) bf16
  __bf16* Asb = (__bf16*)As4;
  __bf16* Bsb = (__bf16*)Bs4;
  const int tid = threadIdx.x, lane = tid & 31, wv = tid >> 5;
  const int rowBase = blockIdx.x * 128;
  const int bh = blockIdx.y;

  FragC acc[4];
  for (int ct = 0; ct < 4; ++ct)
    for (int g = 0; g < 8; ++g) acc[ct].f[g] = 0.f;

  for (int k0 = 0; k0 < kL / 32; ++k0) {
    // stage A: prob f32 -> bf16
    {
      int row = tid >> 1, half = tid & 1;
      const float4* src = (const float4*)(P +
          ((size_t)bh * kT + rowBase + row) * kL + k0 * 32 + half * 16);
      for (int j = 0; j < 4; ++j) {
        float4 f = src[j];
        int o = row * 32 + half * 16 + j * 4;
        Asb[o + 0] = (__bf16)f.x;
        Asb[o + 1] = (__bf16)f.y;
        Asb[o + 2] = (__bf16)f.z;
        Asb[o + 3] = (__bf16)f.w;
      }
    }
    // stage B transposed: Bs[e][kk] = Vcat[(k0*32+kk)*64 + e]
    {
      int kkk = tid >> 3, e0 = (tid & 7) * 8;
      FragAB u;
      u.q[0] = *((const uint4*)(Vcat +
          ((size_t)bh * kL + k0 * 32 + kkk) * kE + e0));
      const __bf16* eight = (const __bf16*)&u.q[0];
      for (int j = 0; j < 8; ++j)
        Bsb[(e0 + j) * 32 + kkk] = eight[j];
    }
    __syncthreads();
    bf16x16 a = load_fragA(As4, wv * 16, 4, 0, lane);
    for (int ct = 0; ct < 4; ++ct) {
      bf16x16 b = load_fragB(Bs4, ct * 16, 4, 0, lane);
      acc[ct].v = wmma_bf16(a, b, acc[ct].v);
    }
    __syncthreads();
  }

  const int half = lane >> 4, n = lane & 15;
  for (int ct = 0; ct < 4; ++ct)
    for (int g = 0; g < 8; ++g) {
      int i = rowBase + wv * 16 + g + 8 * half;
      int e = ct * 16 + n;
      VECf[((size_t)bh * kT + i) * kE + e] = acc[ct].f[g];
    }
}

// vec [b][h][t][e] f32  ->  (t*B+b, h*64+e) bf16 row-major
__global__ void vec_reorder_kernel(const float* __restrict__ VECf,
                                   __bf16* __restrict__ VECbf) {
  size_t idx = (size_t)blockIdx.x * 256 + threadIdx.x;  // < 2048*512
  int d = (int)(idx & 511);
  int r = (int)(idx >> 9);
  int t = r >> 2, b = r & 3;
  int h = d >> 6, e = d & 63;
  VECbf[idx] = (__bf16)VECf[(((size_t)(b * kH + h)) * kT + t) * kE + e];
}

// residual + LayerNorm, one block per row of (T*B, D)
__global__ __launch_bounds__(256)
void ln_kernel(const float* __restrict__ x, const float* __restrict__ aout,
               const float* __restrict__ g, const float* __restrict__ beta,
               float* __restrict__ out) {
  const int r = blockIdx.x, tid = threadIdx.x;
  __shared__ float red[256];
  const float* xr = x + (size_t)r * kD;
  const float* ar = aout + (size_t)r * kD;
  float v0 = xr[tid] + ar[tid];
  float v1 = xr[tid + 256] + ar[tid + 256];
  red[tid] = v0 + v1; __syncthreads();
  for (int s = 128; s > 0; s >>= 1) {
    if (tid < s) red[tid] += red[tid + s];
    __syncthreads();
  }
  float mu = red[0] * (1.f / kD); __syncthreads();
  float d0 = v0 - mu, d1 = v1 - mu;
  red[tid] = d0 * d0 + d1 * d1; __syncthreads();
  for (int s = 128; s > 0; s >>= 1) {
    if (tid < s) red[tid] += red[tid + s];
    __syncthreads();
  }
  float inv = rsqrtf(red[0] * (1.f / kD) + 1e-5f);
  out[(size_t)r * kD + tid]       = d0 * inv * g[tid] + beta[tid];
  out[(size_t)r * kD + tid + 256] = d1 * inv * g[tid + 256] + beta[tid + 256];
}

// ---------------------------------------------------------------------------
// Host launcher
// ---------------------------------------------------------------------------
extern "C" void kernel_launch(void* const* d_in, const int* in_sizes, int n_in,
                              void* d_out, int out_size, void* d_ws,
                              size_t ws_size, hipStream_t stream) {
  const float* x        = (const float*)d_in[0];
  const float* pos_emb  = (const float*)d_in[1];
  const float* pbu      = (const float*)d_in[2];
  const float* pbv      = (const float*)d_in[3];
  const unsigned char* mask = (const unsigned char*)d_in[4];
  const float* memory   = (const float*)d_in[5];
  const float* indice   = (const float*)d_in[6];
  const float* weights  = (const float*)d_in[7];
  const float* nei      = (const float*)d_in[8];
  const float* Wq       = (const float*)d_in[9];
  const float* Wkv      = (const float*)d_in[10];
  const float* Wrel     = (const float*)d_in[11];
  const float* Wo       = (const float*)d_in[12];
  const float* ln_g     = (const float*)d_in[13];
  const float* ln_b     = (const float*)d_in[14];

  float* out      = (float*)d_out;                       // (T,B,D)
  float* out_attn = out + (size_t)kT * kB * kD;          // (T,L)

  // workspace carve-out
  char* ws = (char*)d_ws;
  size_t off = 0;
  auto alloc = [&](size_t bytes) {
    size_t r = off;
    off += (bytes + 255) & ~(size_t)255;
    return r;
  };
  const size_t RC   = (size_t)(kM * kT + kT) * kB;   // 10240 rows of c
  const size_t RPOS = (size_t)kL * kB;               // 12288 rows of pos
  const size_t RX   = (size_t)kT * kB;               // 2048 rows of x / nei

  __bf16* Cbf   = (__bf16*)(ws + alloc(RC * kD * 2));
  __bf16* POSbf = (__bf16*)(ws + alloc(RPOS * kD * 2));
  __bf16* NEIbf = (__bf16*)(ws + alloc(RX * kD * 2));
  __bf16* WkvT  = (__bf16*)(ws + alloc((size_t)2 * kD * kD * 2));
  __bf16* WqT   = (__bf16*)(ws + alloc((size_t)kD * kD * 2));
  __bf16* WrelT = (__bf16*)(ws + alloc((size_t)kD * kD * 2));
  __bf16* WoT   = (__bf16*)(ws + alloc((size_t)kD * kD * 2));
  float*  KV    = (float*)(ws + alloc(RC * 2 * kD * 4));
  float*  RELf  = (float*)(ws + alloc(RPOS * kD * 4));
  float*  Qf    = (float*)(ws + alloc(RX * kD * 4));
  float*  NKV   = (float*)(ws + alloc(RX * 2 * kD * 4));
  __bf16* QU    = (__bf16*)(ws + alloc((size_t)kB * kH * kT * kE * 2));
  __bf16* QV    = (__bf16*)(ws + alloc((size_t)kB * kH * kT * kE * 2));
  __bf16* Kcat  = (__bf16*)(ws + alloc((size_t)kB * kH * kL * kE * 2));
  __bf16* Rcat  = (__bf16*)(ws + alloc((size_t)kB * kH * kL * kE * 2));
  __bf16* Vcat  = (__bf16*)(ws + alloc((size_t)kB * kH * kL * kE * 2));
  float*  SCORE = (float*)(ws + alloc((size_t)kB * kH * kT * kL * 4));
  float*  VECf  = (float*)(ws + alloc((size_t)kB * kH * kT * kE * 4));
  __bf16* VECbf = (__bf16*)(ws + alloc(RX * kD * 2));
  float*  AOUT  = (float*)(ws + alloc(RX * kD * 4));
  (void)ws_size; (void)in_sizes; (void)n_in; (void)out_size;

  const int thr = 256;
  const int gemmLds  = 16384;   // 8KB A + 8KB B
  const int scoreLds = 40960;   // 8KB Q + 32KB K/R

  // --- precision conversion / layout prep ---
  build_cbf_kernel<<<(unsigned)(RC * kD / thr), thr, 0, stream>>>(memory, x, Cbf);
  f2bf_kernel<<<(unsigned)(RPOS * kD / thr), thr, 0, stream>>>(pos_emb, POSbf, RPOS * kD);
  f2bf_kernel<<<(unsigned)(RX * kD / thr), thr, 0, stream>>>(nei, NEIbf, RX * kD);
  transpose_bf_kernel<<<(unsigned)((size_t)kD * 2 * kD / thr), thr, 0, stream>>>(Wkv, WkvT, kD, 2 * kD);
  transpose_bf_kernel<<<(unsigned)((size_t)kD * kD / thr), thr, 0, stream>>>(Wq, WqT, kD, kD);
  transpose_bf_kernel<<<(unsigned)((size_t)kD * kD / thr), thr, 0, stream>>>(Wrel, WrelT, kD, kD);
  transpose_bf_kernel<<<(unsigned)((size_t)kD * kD / thr), thr, 0, stream>>>(Wo, WoT, kD, kD);

  // --- projection GEMMs (bf16 WMMA, async LDS staging) ---
  gemm_bf16_kernel<<<dim3(RC / 128, (2 * kD) / 128), thr, gemmLds, stream>>>(Cbf, WkvT, KV, kD, 2 * kD);
  gemm_bf16_kernel<<<dim3(RPOS / 128, kD / 128), thr, gemmLds, stream>>>(POSbf, WrelT, RELf, kD, kD);
  gemm_bf16_kernel<<<dim3(RX / 128, kD / 128), thr, gemmLds, stream>>>(Cbf + (size_t)kM * kT * kB * kD, WqT, Qf, kD, kD);
  gemm_bf16_kernel<<<dim3(RX / 128, (2 * kD) / 128), thr, gemmLds, stream>>>(NEIbf, WkvT, NKV, kD, 2 * kD);

  // --- attention input prep ---
  build_qubf_kernel<<<(unsigned)((size_t)kB * kH * kT * kE / thr), thr, 0, stream>>>(Qf, pbu, pbv, QU, QV);
  build_kcat_kernel<<<(unsigned)((size_t)kB * kH * kL * kE / thr), thr, 0, stream>>>(KV, NKV, RELf, Kcat, Rcat);
  build_vcat_kernel<<<(unsigned)((size_t)kB * kH * kL * kE / thr), thr, 0, stream>>>(KV, NKV, weights, Vcat);

  // --- scores (WMMA), rel_shift folded into BD scatter-add ---
  score_ac_kernel<<<dim3(kL / 256, kT / 64, kB * kH), thr, scoreLds, stream>>>(QU, Kcat, indice, SCORE);
  score_bd_kernel<<<dim3(kL / 256, kT / 64, kB * kH), thr, scoreLds, stream>>>(QV, Rcat, indice, SCORE);

  // --- softmax + attn matrix ---
  softmax_kernel<<<kB * kH * kT, thr, 0, stream>>>(SCORE, mask);
  attnmat_kernel<<<(unsigned)((size_t)kT * kL / thr), thr, 0, stream>>>(SCORE, out_attn);

  // --- prob @ V (WMMA) ---
  pv_kernel<<<dim3(kT / 128, kB * kH), thr, 0, stream>>>(SCORE, Vcat, VECf);
  vec_reorder_kernel<<<(unsigned)(RX * kD / thr), thr, 0, stream>>>(VECf, VECbf);

  // --- output projection + residual LayerNorm ---
  gemm_bf16_kernel<<<dim3(RX / 128, kD / 128), thr, gemmLds, stream>>>(VECbf, WoT, AOUT, kD, kD);
  ln_kernel<<<RX, thr, 0, stream>>>(x, AOUT, ln_g, ln_b, out);
}
